// SwinTransformer_26061861552864
// MI455X (gfx1250) — matrix operand order repaired and checked by reading, not compiled, not run on hardware
//
#include <hip/hip_runtime.h>
#include <hip/hip_bf16.h>

// ---------------------------------------------------------------------------
// Swin-T forward for MI455X (gfx1250, wave32, WMMA + TDM).
// f16 activation storage (HBM-bound model), fp32 residual stream.
// GEMM A-tiles are DMA'd global->LDS by the Tensor Data Mover (double
// buffered, TENSORcnt-tracked); all GEMMs use v_wmma_f32_16x16x32_f16.
// ---------------------------------------------------------------------------

typedef _Float16 h16;
typedef __attribute__((ext_vector_type(16))) _Float16 v16h;
typedef __attribute__((ext_vector_type(8)))  _Float16 v8h;
typedef __attribute__((ext_vector_type(4)))  _Float16 v4h;
typedef __attribute__((ext_vector_type(8)))  float    v8f;
typedef __attribute__((ext_vector_type(4)))  float    v4f;
typedef __attribute__((ext_vector_type(4)))  unsigned int v4u;
typedef __attribute__((ext_vector_type(8)))  int      v8i;
typedef __attribute__((ext_vector_type(4)))  int      v4i;

#define WMMA_F16(a, b, c) \
  __builtin_amdgcn_wmma_f32_16x16x32_f16(false, (a), false, (b), (short)0, (c), false, false)

// ---- WMMA fragment loaders (CDNA5 16-bit layouts, CDNA5 ISA) ----
__device__ inline v16h frag_a(const h16* base, int lane) {
  int k0 = (lane >> 4) * 8;
  v8h lo = *(const v8h*)(base + k0);
  v8h hi = *(const v8h*)(base + 16 + k0);
  v16h r;
#pragma unroll
  for (int i = 0; i < 8; ++i) { r[i] = lo[i]; r[i + 8] = hi[i]; }
  return r;
}
__device__ inline v16h frag_b(const h16* base, int lane) {
  int k0 = (lane >> 4) * 8;
  v8h lo = *(const v8h*)(base + k0);
  v8h hi = *(const v8h*)(base + 16 + k0);
  v16h r;
#pragma unroll
  for (int i = 0; i < 8; ++i) { r[i] = lo[i]; r[i + 8] = hi[i]; }
  return r;
}

// ---------------------------------------------------------------------------
// TDM: DMA a 128x32-half A tile (f16, row stride K halves) into LDS with
// row padding 32->40 halves (pad_interval=16 dwords, pad_amount=4 dwords).
// D# packing per CDNA5 ISA (2D tensor; groups 2/3 and
// the trailing 256-bit group unused -> zero). 6-arg builtin (clang-23 form).
// ---------------------------------------------------------------------------
__device__ __forceinline__ void tdm_load_a_tile(const h16* gsrc, unsigned ldsOff,
                                                int K, int rowsLeft) {
  unsigned long long ga = (unsigned long long)gsrc;
  v4u g0;
  g0[0] = 1u;                                              // count=1, user D#
  g0[1] = ldsOff;                                          // lds_addr (bytes)
  g0[2] = (unsigned)ga;                                    // global_addr[31:0]
  g0[3] = (unsigned)((ga >> 32) & 0x01ffffffu) | (2u << 30); // ga[56:32], type=2
  unsigned td0 = (unsigned)K;                              // tensor_dim0 (elems)
  unsigned td1 = (unsigned)rowsLeft;                       // tensor_dim1 (rows)
  v8i g1;
  g1[0] = (int)((1u << 16)        // data_size = 1 -> 2 bytes
              | (1u << 20)        // pad_enable
              | (3u << 22)        // pad_interval: 16 dwords (one 32-half row)
              | (3u << 25));      // pad_amount: 4 dwords (8 halves)
  g1[1] = (int)((td0 & 0xffffu) << 16);                       // tensor_dim0 lo
  g1[2] = (int)(((td0 >> 16) & 0xffffu) | ((td1 & 0xffffu) << 16));
  g1[3] = (int)(((td1 >> 16) & 0xffffu) | (32u << 16));       // tile_dim0 = 32
  g1[4] = (int)128u;                                          // tile_dim1 = 128
  g1[5] = (int)td0;                                           // dim0_stride lo
  g1[6] = 0;                                                  // dim0_stride hi
  g1[7] = 0;
  v4i gz4 = {0, 0, 0, 0};
  v8i gz8 = {0, 0, 0, 0, 0, 0, 0, 0};
  __builtin_amdgcn_tensor_load_to_lds(g0, g1, gz4, gz4, gz8, 0);
}

// ---------------------------------------------------------------------------
// GEMM: out[M,N] = act(A[M,K] @ W[N,K]^T + bias) (+ residual)
// A f16 row-major (TDM-staged, double buffered), W fp32 (torch Linear).
// K % 32 == 0. 256 threads (8 waves); tile 128x64x32; 32 WMMAs / k-step.
// ---------------------------------------------------------------------------
__global__ __launch_bounds__(256) void gemm_wmma(
    const h16* __restrict__ A, const float* __restrict__ W,
    const float* __restrict__ bias, const float* __restrict__ res,
    void* __restrict__ CoutV, int M, int N, int K, int act, int outF16) {
  __shared__ h16 aLds[2 * 128 * 40];  // ping-pong [m][k], stride 40 halves
  __shared__ h16 bLds[64 * 40];       // [n][k]
  const int tid = threadIdx.x, lane = tid & 31, wave = tid >> 5;
  const int mBlock = blockIdx.y * 128, nBlock = blockIdx.x * 64;
  const int steps = K >> 5;
  const h16* aBase = A + (long long)mBlock * K;
  const int rowsLeft = M - mBlock;

  if (wave == 0)
    tdm_load_a_tile(aBase, (unsigned)(unsigned long long)&aLds[0], K, rowsLeft);

  v8f acc[4] = {v8f{}, v8f{}, v8f{}, v8f{}};
  for (int s = 0; s < steps; ++s) {
    const int k0 = s << 5;
    const int cur = s & 1;
    if (wave == 0 && s + 1 < steps)
      tdm_load_a_tile(aBase + k0 + 32,
                      (unsigned)(unsigned long long)&aLds[(cur ^ 1) * 128 * 40],
                      K, rowsLeft);
    // ---- B tile: 64x32 floats = 512 float4 chunks; 2 per thread ----
#pragma unroll
    for (int it = 0; it < 2; ++it) {
      int chunk = tid + it * 256;          // chunk = n*8 + c4
      int n = chunk >> 3, c0 = (chunk & 7) * 4;
      int gn = nBlock + n;
      v4h h;
      if (gn < N) {
        v4f v = *(const v4f*)(W + (long long)gn * K + k0 + c0);
        if (k0 + 32 < K) __builtin_prefetch(W + (long long)gn * K + k0 + 32 + c0, 0, 3);
#pragma unroll
        for (int i = 0; i < 4; ++i) h[i] = (h16)v[i];
      } else {
        h = v4h{};
      }
      *(v4h*)(&bLds[n * 40 + c0]) = h;
    }
    if (wave == 0) {
      if (s + 1 < steps) __builtin_amdgcn_s_wait_tensorcnt(1);  // current tile done
      else               __builtin_amdgcn_s_wait_tensorcnt(0);
    }
    __syncthreads();
    const h16* aBuf = &aLds[cur * 128 * 40];
    v16h a = frag_a(&aBuf[(wave * 16 + (lane & 15)) * 40], lane);
#pragma unroll
    for (int nt = 0; nt < 4; ++nt) {
      v16h b = frag_b(&bLds[(nt * 16 + (lane & 15)) * 40], lane);
      acc[nt] = WMMA_F16(a, b, acc[nt]);
    }
    __syncthreads();
  }
  const int hi = lane >> 4;
#pragma unroll
  for (int nt = 0; nt < 4; ++nt) {
    int gn = nBlock + nt * 16 + (lane & 15);
    if (gn >= N) continue;
    float bv = bias ? bias[gn] : 0.f;
#pragma unroll
    for (int e = 0; e < 8; ++e) {
      int gm = mBlock + wave * 16 + e + 8 * hi;
      if (gm >= M) continue;
      float v = acc[nt][e] + bv;
      if (act == 1) v = 0.5f * v * (1.0f + erff(v * 0.7071067811865475f));
      if (res) v += res[(long long)gm * N + gn];
      if (outF16) ((h16*)CoutV)[(long long)gm * N + gn] = (h16)v;
      else        ((float*)CoutV)[(long long)gm * N + gn] = v;
    }
  }
}

// ---------------------------------------------------------------------------
// Patch embed: conv 4x4 stride 4 -> [B, 56*56, 96] (fp32 temp)
// ---------------------------------------------------------------------------
__global__ void patch_embed(const float* __restrict__ x, const float* __restrict__ w,
                            const float* __restrict__ b, float* __restrict__ out, int B) {
  long long i = (long long)blockIdx.x * blockDim.x + threadIdx.x;
  long long total = (long long)B * 56 * 56 * 96;
  if (i >= total) return;
  int oc = (int)(i % 96); long long t = i / 96;
  int ox = (int)(t % 56); long long t2 = t / 56;
  int oy = (int)(t2 % 56); int bb = (int)(t2 / 56);
  float acc = b[oc];
#pragma unroll
  for (int ic = 0; ic < 3; ++ic)
#pragma unroll
    for (int ky = 0; ky < 4; ++ky)
#pragma unroll
      for (int kx = 0; kx < 4; ++kx) {
        float xv = x[(((long long)bb * 3 + ic) * 224 + (oy * 4 + ky)) * 224 + (ox * 4 + kx)];
        float wv = w[((oc * 3 + ic) * 4 + ky) * 4 + kx];
        acc += xv * wv;
      }
  out[i] = acc;
}

// ---------------------------------------------------------------------------
// LayerNorm: one wave per row (wave32). Output type templated (fp32 or f16).
// ---------------------------------------------------------------------------
template <typename OT>
__global__ __launch_bounds__(256) void layernorm(
    const float* __restrict__ x, const float* __restrict__ w, const float* __restrict__ b,
    OT* __restrict__ y, int T, int C) {
  int row = blockIdx.x * (blockDim.x >> 5) + (threadIdx.x >> 5);
  int lane = threadIdx.x & 31;
  if (row >= T) return;
  const float* xr = x + (long long)row * C;
  float s = 0.f, s2 = 0.f;
  for (int c = lane; c < C; c += 32) { float v = xr[c]; s += v; s2 += v * v; }
#pragma unroll
  for (int o = 16; o > 0; o >>= 1) { s += __shfl_xor(s, o, 32); s2 += __shfl_xor(s2, o, 32); }
  float mu = s / C;
  float inv = rsqrtf(s2 / C - mu * mu + 1e-6f);
  OT* yr = y + (long long)row * C;
  for (int c = lane; c < C; c += 32) yr[c] = (OT)((xr[c] - mu) * inv * w[c] + b[c]);
}

// ---------------------------------------------------------------------------
// Windowed attention, one block per (window, head). head_dim == 32.
// QKV f16 in, O f16 out. Roll folded into gather/scatter indices.
// ---------------------------------------------------------------------------
__device__ inline int regionOf(int r, int D, int shift) {
  if (r < D - 7) return 0;
  if (r < D - shift) return 1;
  return 2;
}

__global__ __launch_bounds__(128) void win_attn(
    const h16* __restrict__ qkv,    // [T, 3C] f16
    const float* __restrict__ rpb,  // [169, nH]
    h16* __restrict__ O,            // [T, C] f16
    int H, int Wd, int C, int nH, int shift) {
  __shared__ h16 qLds[64 * 40];
  __shared__ h16 kLds[64 * 40];
  __shared__ h16 vtLds[32 * 72];
  __shared__ float sLds[64 * 65];
  __shared__ h16 pLds[64 * 72];
  __shared__ int rowIdx[49];

  const int tid = threadIdx.x, lane = tid & 31, wave = tid >> 5;
  const int head = blockIdx.y;
  const int win = blockIdx.x;
  const int nWw = Wd / 7, nWh = H / 7, wpi = nWh * nWw;
  const int bb = win / wpi, wrem = win % wpi;
  const int wr = wrem / nWw, wc = wrem % nWw;

  for (int t = tid; t < 49; t += 128) {
    int lr = t / 7, lc = t % 7;
    int sr = (wr * 7 + lr + shift) % H;
    int sc = (wc * 7 + lc + shift) % Wd;
    rowIdx[t] = (bb * H + sr) * Wd + sc;
  }
  __syncthreads();

  const float scale = 0.17677669529663689f;  // 1/sqrt(32)
#pragma unroll
  for (int it = 0; it < 2; ++it) {
    int idx = tid + it * 128;
    int m = idx >> 2, d0 = (idx & 3) * 8;
    v8h qv, kv, vv;
    if (m < 49) {
      const h16* rp = qkv + (long long)rowIdx[m] * 3 * C + head * 32 + d0;
      qv = *(const v8h*)(rp);
      kv = *(const v8h*)(rp + C);
      vv = *(const v8h*)(rp + 2 * C);
#pragma unroll
      for (int i = 0; i < 8; ++i) qv[i] = (h16)((float)qv[i] * scale);
    } else {
      qv = v8h{}; kv = v8h{}; vv = v8h{};
    }
    *(v8h*)(&qLds[m * 40 + d0]) = qv;
    *(v8h*)(&kLds[m * 40 + d0]) = kv;
#pragma unroll
    for (int i = 0; i < 8; ++i) vtLds[(d0 + i) * 72 + m] = vv[i];
  }
  __syncthreads();

  // ---- S = Q @ K^T + bias (+mask) ----
  {
    v16h a = frag_a(&qLds[(wave * 16 + (lane & 15)) * 40], lane);
#pragma unroll
    for (int nt = 0; nt < 4; ++nt) {
      v16h b = frag_b(&kLds[(nt * 16 + (lane & 15)) * 40], lane);
      v8f c = {};
      c = WMMA_F16(a, b, c);
      int n = nt * 16 + (lane & 15);
#pragma unroll
      for (int e = 0; e < 8; ++e) {
        int m = wave * 16 + e + 8 * (lane >> 4);
        float v;
        if (m < 49 && n < 49) {
          int ri = m / 7, ci = m % 7, rj = n / 7, cj = n % 7;
          int idxr = (ri - rj + 6) * 13 + (ci - cj + 6);
          v = c[e] + rpb[idxr * nH + head];
          if (shift) {
            int rm = regionOf(wr * 7 + ri, H, shift) * 3 + regionOf(wc * 7 + ci, Wd, shift);
            int rn = regionOf(wr * 7 + rj, H, shift) * 3 + regionOf(wc * 7 + cj, Wd, shift);
            if (rm != rn) v -= 100.0f;
          }
        } else {
          v = -1e30f;
        }
        sLds[m * 65 + n] = v;
      }
    }
  }
  __syncthreads();

  // ---- row softmax, emit f16 probs padded to K=64 ----
  if (tid < 64) {
    int m = tid;
    if (m < 49) {
      float mx = -1e30f;
      for (int n = 0; n < 49; ++n) mx = fmaxf(mx, sLds[m * 65 + n]);
      float sum = 0.f;
      for (int n = 0; n < 49; ++n) { float e = __expf(sLds[m * 65 + n] - mx); sLds[m * 65 + n] = e; sum += e; }
      float inv = 1.f / sum;
      for (int n = 0; n < 64; ++n) pLds[m * 72 + n] = (h16)((n < 49) ? sLds[m * 65 + n] * inv : 0.f);
    } else {
      for (int n = 0; n < 64; ++n) pLds[m * 72 + n] = (h16)0.f;
    }
  }
  __syncthreads();

  // ---- out = P @ V  (K = 64, two steps) ----
  v8f oacc[2] = {v8f{}, v8f{}};
#pragma unroll
  for (int ks = 0; ks < 2; ++ks) {
    v16h a = frag_a(&pLds[(wave * 16 + (lane & 15)) * 72 + ks * 32], lane);
#pragma unroll
    for (int nt = 0; nt < 2; ++nt) {
      v16h b = frag_b(&vtLds[(nt * 16 + (lane & 15)) * 72 + ks * 32], lane);
      oacc[nt] = WMMA_F16(a, b, oacc[nt]);
    }
  }
#pragma unroll
  for (int nt = 0; nt < 2; ++nt) {
    int d = nt * 16 + (lane & 15);
#pragma unroll
    for (int e = 0; e < 8; ++e) {
      int m = wave * 16 + e + 8 * (lane >> 4);
      if (m < 49) O[(long long)rowIdx[m] * C + head * 32 + d] = (h16)oacc[nt][e];
    }
  }
}

// ---------------------------------------------------------------------------
// Patch-merge gather: [B,H,W,C] -> [B,(H/2)(W/2),4C] fp32
// ---------------------------------------------------------------------------
__global__ void merge_gather(const float* __restrict__ x, float* __restrict__ y,
                             int B, int H, int Wd, int C) {
  long long i = (long long)blockIdx.x * blockDim.x + threadIdx.x;
  int Ho = H / 2, Wo = Wd / 2;
  long long total = (long long)B * Ho * Wo * 4 * C;
  if (i >= total) return;
  int c4 = (int)(i % (4 * C)); long long t = i / (4 * C);
  int ox = (int)(t % Wo); long long t2 = t / Wo;
  int oy = (int)(t2 % Ho); int bb = (int)(t2 / Ho);
  int group = c4 / C, c = c4 % C;
  int dy = (group == 1 || group == 3) ? 1 : 0;
  int dx = (group >= 2) ? 1 : 0;
  y[i] = x[(((long long)bb * H + oy * 2 + dy) * Wd + ox * 2 + dx) * C + c];
}

// ---------------------------------------------------------------------------
// Host orchestration
// ---------------------------------------------------------------------------
static const int DEPTHS[4] = {2, 2, 6, 2};
static const int HEADS_[4] = {3, 6, 12, 24};

extern "C" void kernel_launch(void* const* d_in, const int* in_sizes, int n_in,
                              void* d_out, int out_size, void* d_ws, size_t ws_size,
                              hipStream_t stream) {
  (void)in_sizes; (void)n_in; (void)out_size; (void)ws_size;
  // JAX pytree flatten order (dict keys sorted, lists in order):
  //  0 conv_b, 1 conv_w, 2..5 out_nb[0..3], 6..9 out_nw[0..3], 10 pe_nb, 11 pe_nw,
  //  stages base {12,41,70,151}; per block (13): fc1b,fc1w,fc2b,fc2w,n1b,n1w,
  //  n2b,n2w,projb,projw,qkvb,qkvw,rpb; merge (stages 0..2): nb,nw,redw. x = 177.
  auto IN = [&](int i) -> const float* { return (const float*)d_in[i]; };
  static const int stage_base[4] = {12, 41, 70, 151};

  const int B = 16;
  float* ws = (float*)d_ws;
  const long long SZ = 4816896LL;  // max T*C = 16*3136*96
  float* X    = ws;                       // fp32 residual stream [T,C]
  float* Yf   = ws + SZ;                  // fp32 temp (patch embed / merge gather)
  h16*   Yh   = (h16*)Yf;                 // f16 LN output [T,C]
  h16*   Oh   = (h16*)(ws + 2 * SZ);      // f16 attn out / merge LN out
  h16*   QKVh = (h16*)(ws + 3 * SZ);      // f16 [T,3C]
  h16*   HBh  = (h16*)(ws + 6 * SZ);      // f16 [T,4C]
  const float* x_img = IN(177);

  long long out_off[4];
  out_off[0] = 0;
  out_off[1] = out_off[0] + (long long)B * 3136 * 96;
  out_off[2] = out_off[1] + (long long)B * 784 * 192;
  out_off[3] = out_off[2] + (long long)B * 196 * 384;

  // patch embed (fp32 temp) + embedding LN (fp32 residual stream)
  {
    long long total = (long long)B * 56 * 56 * 96;
    patch_embed<<<dim3((unsigned)((total + 255) / 256)), 256, 0, stream>>>(
        x_img, IN(1), IN(0), Yf, B);
    int T = B * 3136;
    layernorm<float><<<dim3((T + 7) / 8), 256, 0, stream>>>(Yf, IN(11), IN(10), X, T, 96);
  }

  int H = 56, Wd = 56;
  for (int i = 0; i < 4; ++i) {
    const int C = 96 << i, nH = HEADS_[i], T = B * H * Wd;
    const int nWin = B * (H / 7) * (Wd / 7);
    for (int d = 0; d < DEPTHS[i]; ++d) {
      const int p = stage_base[i] + d * 13;
      const float *fc1b = IN(p+0), *fc1w = IN(p+1), *fc2b = IN(p+2), *fc2w = IN(p+3);
      const float *n1b = IN(p+4), *n1w = IN(p+5), *n2b = IN(p+6), *n2w = IN(p+7);
      const float *projb = IN(p+8), *projw = IN(p+9), *qkvb = IN(p+10), *qkvw = IN(p+11);
      const float *rpb = IN(p+12);
      const int shift = (d % 2 == 1) ? 3 : 0;

      layernorm<h16><<<dim3((T + 7) / 8), 256, 0, stream>>>(X, n1w, n1b, Yh, T, C);
      gemm_wmma<<<dim3((3 * C + 63) / 64, (T + 127) / 128), 256, 0, stream>>>(
          Yh, qkvw, qkvb, nullptr, QKVh, T, 3 * C, C, 0, 1);
      win_attn<<<dim3(nWin, nH), 128, 0, stream>>>(QKVh, rpb, Oh, H, Wd, C, nH, shift);
      gemm_wmma<<<dim3((C + 63) / 64, (T + 127) / 128), 256, 0, stream>>>(
          Oh, projw, projb, X, X, T, C, C, 0, 0);
      layernorm<h16><<<dim3((T + 7) / 8), 256, 0, stream>>>(X, n2w, n2b, Yh, T, C);
      gemm_wmma<<<dim3((4 * C + 63) / 64, (T + 127) / 128), 256, 0, stream>>>(
          Yh, fc1w, fc1b, nullptr, HBh, T, 4 * C, C, 1, 1);
      gemm_wmma<<<dim3((C + 63) / 64, (T + 127) / 128), 256, 0, stream>>>(
          HBh, fc2w, fc2b, X, X, T, C, 4 * C, 0, 0);
    }
    layernorm<float><<<dim3((T + 7) / 8), 256, 0, stream>>>(
        X, IN(6 + i), IN(2 + i), (float*)d_out + out_off[i], T, C);
    if (i < 3) {
      const int mp = stage_base[i] + DEPTHS[i] * 13;  // nb, nw, redw
      long long total = (long long)T * C;
      merge_gather<<<dim3((unsigned)((total + 255) / 256)), 256, 0, stream>>>(
          X, Yf, B, H, Wd, C);
      layernorm<h16><<<dim3((T / 4 + 7) / 8), 256, 0, stream>>>(
          Yf, IN(mp + 1), IN(mp + 0), Oh, T / 4, 4 * C);
      gemm_wmma<<<dim3((2 * C + 63) / 64, (T / 4 + 127) / 128), 256, 0, stream>>>(
          Oh, IN(mp + 2), nullptr, nullptr, X, T / 4, 2 * C, 4 * C, 0, 0);
      H >>= 1; Wd >>= 1;
    }
  }
}
